// GraphAutoencoder_33981781246138
// MI455X (gfx1250) — matrix operand order repaired and checked by reading, not compiled, not run on hardware
//
#include <hip/hip_runtime.h>

typedef __attribute__((ext_vector_type(16))) _Float16 v16h;
typedef __attribute__((ext_vector_type(8)))  _Float16 v8h;
typedef __attribute__((ext_vector_type(4)))  _Float16 v4h;
typedef __attribute__((ext_vector_type(2)))  _Float16 v2h;
typedef __attribute__((ext_vector_type(2)))  __fp16   v2fp;
typedef __attribute__((ext_vector_type(8)))  float    v8f;

#define EMB      16
#define LAT      8
#define EDGE_DIM 32
#define HID      64
#define THETA    (EMB * LAT)   // 128
#define EPT      32            // edges per wave-tile (two 16-row M subtiles)

// packed f32->f16 convert (v_cvt_pk_rtz_f16_f32), bit-cast __fp16x2 -> _Float16x2
__device__ __forceinline__ v2h cvt_pk_h2(float x, float y) {
    v2fp t = __builtin_amdgcn_cvt_pkrtz(x, y);
    return __builtin_bit_cast(v2h, t);
}

// ---------------------------------------------------------------- fragments
// B (Kx16 f16): col N = lane&15; b0 = 8*(lane>=16); halves 0-7 -> K = b0 + h,
// halves 8-15 -> K = 16 + b0 + (h-8) (same K order as load_a32).
// Pre-shuffled storage: frag[fragIdx][lane][h], 16 contiguous halves per lane.
__device__ __forceinline__ v16h load_frag(const _Float16* __restrict__ base,
                                          int frag, int lane) {
    const v8h* p = (const v8h*)(base + (((size_t)frag * 32 + lane) << 4));
    v8h lo = p[0], hi = p[1];
    return __builtin_shufflevector(lo, hi, 0,1,2,3,4,5,6,7,8,9,10,11,12,13,14,15);
}

// A (16x32 f16): lane m=lane&15; halves 0-7 = K[b0..b0+7], halves 8-15 =
// K[16+b0..16+b0+7], b0 = 8*(lane>=16)  -> two contiguous 16B runs.
__device__ __forceinline__ v16h load_a32(const _Float16* rowPtr, int lane) {
    int b0 = (lane >> 4) << 3;
    v8h lo = *(const v8h*)(rowPtr + b0);
    v8h hi = *(const v8h*)(rowPtr + 16 + b0);
    return __builtin_shufflevector(lo, hi, 0,1,2,3,4,5,6,7,8,9,10,11,12,13,14,15);
}

// C/D (16x16 f32): col N = lane&15; row M = r + 8*(lane>=16).
__device__ __forceinline__ int cd_m(int lane, int r) {
    return r + ((lane >> 4) << 3);
}

// ------------------------------------------------------------------ zero ws
__global__ void zero_kernel(float* __restrict__ agg, float* __restrict__ cnt,
                            int na, int nc) {
    int t = blockIdx.x * blockDim.x + threadIdx.x;
    if (t < na) agg[t] = 0.0f;
    if (t < nc) cnt[t] = 0.0f;
}

// ------------------------------------- weights fp32 -> f16 fragment layout
// 28 fragments x 512 elements: [0,4)=W1  [4,20)=W2  [20,24)=D1(K-pad) [24,28)=D2
__global__ void build_frags(const float* __restrict__ W1, const float* __restrict__ W2,
                            const float* __restrict__ D1, const float* __restrict__ D2,
                            _Float16* __restrict__ W1f, _Float16* __restrict__ W2f,
                            _Float16* __restrict__ D1f, _Float16* __restrict__ D2f) {
    int t = blockIdx.x * blockDim.x + threadIdx.x;
    if (t >= 28 * 512) return;
    int e = t & 511, f = t >> 9;
    int l = e >> 4, h = e & 15;
    int krel = (h < 8) ? (((l >> 4) << 3) + h) : (16 + ((l >> 4) << 3) + h - 8);       // 0..31 within 32-K chunk
    int n = l & 15;
    if (f < 4) {                                        // ew_W1: 32x64
        W1f[f * 512 + e] = (_Float16)W1[krel * 64 + f * 16 + n];
    } else if (f < 20) {                                // ew_W2: 64x128
        int g = f - 4;
        int k = ((g & 1) << 5) + krel;
        W2f[(size_t)g * 512 + e] = (_Float16)W2[k * 128 + (g >> 1) * 16 + n];
    } else if (f < 24) {                                // dec_W1: 16x64, K->32 pad
        int g = f - 20;
        D1f[g * 512 + e] = (krel < 16) ? (_Float16)D1[krel * 64 + g * 16 + n]
                                       : (_Float16)0.0f;
    } else {                                            // dec_W2: 64x32
        int g = f - 24;
        int k = ((g & 1) << 5) + krel;
        D2f[g * 512 + e] = (_Float16)D2[k * 32 + (g >> 1) * 16 + n];
    }
}

// ------------------------------------------- edge MLP + einsum + scatter-add
// One wave per 32-edge tile; lane L owns edge e0+L for gather/scatter phases.
__global__ __launch_bounds__(32)
void edge_msg_kernel(const int* __restrict__ edge_index,
                     const float* __restrict__ edge_attr,
                     const float* __restrict__ node_emb,
                     const _Float16* __restrict__ W1f, const float* __restrict__ b1,
                     const _Float16* __restrict__ W2f, const float* __restrict__ b2,
                     float* __restrict__ agg, float* __restrict__ cnt, int E) {
    __shared__ __align__(16) _Float16 sAttr[EPT * EDGE_DIM];   // 2 KB
    __shared__ __align__(16) _Float16 sH[EPT * HID];           // 4 KB
    __shared__ __align__(16) float    sTheta[EPT * THETA];     // 16 KB

    const int lane = threadIdx.x;
    const int e0   = blockIdx.x * EPT;

    const int src = edge_index[e0 + lane];
    const int tgt = edge_index[(size_t)E + e0 + lane];

    // stage 32x32 edge_attr tile as f16 (coalesced float4, packed cvt)
    const float4* ea4 = (const float4*)(edge_attr + (size_t)e0 * EDGE_DIM);
    #pragma unroll
    for (int t = 0; t < 8; ++t) {
        int idx = t * 32 + lane;                 // float4 index, 0..255
        float4 f = ea4[idx];
        v2h p0 = cvt_pk_h2(f.x, f.y);
        v2h p1 = cvt_pk_h2(f.z, f.w);
        v4h q  = __builtin_shufflevector(p0, p1, 0, 1, 2, 3);
        *(v4h*)&sAttr[(idx >> 3) * EDGE_DIM + (idx & 7) * 4] = q;
    }
    // x = node_emb[src] for this lane's edge (registers only)
    const float4* nr = (const float4*)(node_emb + (size_t)src * EMB);
    float4 x0 = nr[0], x1 = nr[1], x2 = nr[2], x3 = nr[3];
    __syncthreads();

    const int m = lane & 15;
    const int n = lane & 15;

    // A fragments: two 16-edge M subtiles
    v16h a[2];
    #pragma unroll
    for (int mt = 0; mt < 2; ++mt)
        a[mt] = load_a32(&sAttr[(mt * 16 + m) * EDGE_DIM], lane);

    // GEMM1: (32x32)@(32x64) + bias, ReLU -> sH (f16); B reused across subtiles
    #pragma unroll
    for (int nt = 0; nt < 4; ++nt) {
        v16h b = load_frag(W1f, nt, lane);
        float bias = b1[nt * 16 + n];
        #pragma unroll
        for (int mt = 0; mt < 2; ++mt) {
            v8f c = {};
            c = __builtin_amdgcn_wmma_f32_16x16x32_f16(false, a[mt], false, b,
                                                       (short)0, c, false, false);
            #pragma unroll
            for (int r = 0; r < 8; ++r) {
                float v = c[r] + bias;
                sH[(mt * 16 + cd_m(lane, r)) * HID + nt * 16 + n] =
                    (_Float16)(v > 0.0f ? v : 0.0f);
            }
        }
    }
    __syncthreads();

    // A2 fragments: 32x64 as two K-chunks per subtile
    v16h a2[2][2];
    #pragma unroll
    for (int mt = 0; mt < 2; ++mt)
        #pragma unroll
        for (int ch = 0; ch < 2; ++ch)
            a2[mt][ch] = load_a32(&sH[(mt * 16 + m) * HID + ch * 32], lane);

    // GEMM2: (32x64)@(64x128) + bias -> sTheta (f32)
    #pragma unroll
    for (int nt = 0; nt < 8; ++nt) {
        v8f c[2] = {{}, {}};
        #pragma unroll
        for (int ch = 0; ch < 2; ++ch) {
            v16h b = load_frag(W2f, nt * 2 + ch, lane);
            #pragma unroll
            for (int mt = 0; mt < 2; ++mt)
                c[mt] = __builtin_amdgcn_wmma_f32_16x16x32_f16(false, a2[mt][ch], false, b,
                                                               (short)0, c[mt], false, false);
        }
        float bias = b2[nt * 16 + n];
        #pragma unroll
        for (int mt = 0; mt < 2; ++mt)
            #pragma unroll
            for (int r = 0; r < 8; ++r)
                sTheta[(mt * 16 + cd_m(lane, r)) * THETA + nt * 16 + n] = c[mt][r] + bias;
    }
    __syncthreads();

    // msg[o] = sum_i x[i] * theta[lane][i*LAT+o]  (lane owns edge e0+lane)
    float msg[LAT] = {};
    const float xs[EMB] = {x0.x, x0.y, x0.z, x0.w, x1.x, x1.y, x1.z, x1.w,
                           x2.x, x2.y, x2.z, x2.w, x3.x, x3.y, x3.z, x3.w};
    #pragma unroll
    for (int i = 0; i < EMB; ++i) {
        float4 t0 = *(const float4*)&sTheta[lane * THETA + i * LAT];
        float4 t1 = *(const float4*)&sTheta[lane * THETA + i * LAT + 4];
        float xv = xs[i];
        msg[0] += xv * t0.x; msg[1] += xv * t0.y; msg[2] += xv * t0.z; msg[3] += xv * t0.w;
        msg[4] += xv * t1.x; msg[5] += xv * t1.y; msg[6] += xv * t1.z; msg[7] += xv * t1.w;
    }
    #pragma unroll
    for (int o = 0; o < LAT; ++o)
        atomicAdd(&agg[(size_t)tgt * LAT + o], msg[o]);
    atomicAdd(&cnt[tgt], 1.0f);
}

// ------------------------------------------------ latent = emb@rootW + mean
__global__ __launch_bounds__(256)
void latent_kernel(const float* __restrict__ node_emb,
                   const float* __restrict__ rootW,
                   const float* __restrict__ conv_b,
                   const float* __restrict__ agg,
                   const float* __restrict__ cnt,
                   _Float16* __restrict__ latent_h, int N) {
    __shared__ __align__(16) float sW[EMB * LAT];
    __shared__ float sB[LAT];
    int tid = threadIdx.x;
    if (tid < EMB * LAT) sW[tid] = rootW[tid];
    if (tid < LAT) sB[tid] = conv_b[tid];
    __syncthreads();
    int node = blockIdx.x * blockDim.x + tid;
    if (node >= N) return;

    float c = cnt[node]; c = c > 1.0f ? c : 1.0f;
    float inv = 1.0f / c;
    const float4* ag = (const float4*)(agg + (size_t)node * LAT);
    float4 g0 = ag[0], g1 = ag[1];
    float acc[LAT] = {sB[0] + g0.x * inv, sB[1] + g0.y * inv,
                      sB[2] + g0.z * inv, sB[3] + g0.w * inv,
                      sB[4] + g1.x * inv, sB[5] + g1.y * inv,
                      sB[6] + g1.z * inv, sB[7] + g1.w * inv};
    const float4* nr = (const float4*)(node_emb + (size_t)node * EMB);
    #pragma unroll
    for (int q = 0; q < 4; ++q) {
        float4 xv = nr[q];
        const float xq[4] = {xv.x, xv.y, xv.z, xv.w};
        #pragma unroll
        for (int j = 0; j < 4; ++j) {
            float4 w0 = *(const float4*)&sW[(q * 4 + j) * LAT];
            float4 w1 = *(const float4*)&sW[(q * 4 + j) * LAT + 4];
            acc[0] += xq[j] * w0.x; acc[1] += xq[j] * w0.y;
            acc[2] += xq[j] * w0.z; acc[3] += xq[j] * w0.w;
            acc[4] += xq[j] * w1.x; acc[5] += xq[j] * w1.y;
            acc[6] += xq[j] * w1.z; acc[7] += xq[j] * w1.w;
        }
    }
    v2h p0 = cvt_pk_h2(acc[0], acc[1]);
    v2h p1 = cvt_pk_h2(acc[2], acc[3]);
    v2h p2 = cvt_pk_h2(acc[4], acc[5]);
    v2h p3 = cvt_pk_h2(acc[6], acc[7]);
    v4h lo = __builtin_shufflevector(p0, p1, 0, 1, 2, 3);
    v4h hi = __builtin_shufflevector(p2, p3, 0, 1, 2, 3);
    *(v8h*)&latent_h[(size_t)node * LAT] =
        __builtin_shufflevector(lo, hi, 0, 1, 2, 3, 4, 5, 6, 7);
}

// --------------------------------------------- decoder MLP on edges (WMMA)
__global__ __launch_bounds__(32)
void decode_kernel(const int* __restrict__ edge_index,
                   const _Float16* __restrict__ latent_h,
                   const _Float16* __restrict__ D1f, const float* __restrict__ db1,
                   const _Float16* __restrict__ D2f, const float* __restrict__ db2,
                   float* __restrict__ out, int E) {
    __shared__ __align__(16) _Float16 sIn[EPT * 16];   // 1 KB
    __shared__ __align__(16) _Float16 sH[EPT * HID];   // 4 KB

    const int lane = threadIdx.x;
    const int e0   = blockIdx.x * EPT;

    const int s = edge_index[e0 + lane];
    const int g = edge_index[(size_t)E + e0 + lane];
    *(v8h*)&sIn[lane * 16]     = *(const v8h*)(latent_h + (size_t)s * LAT);
    *(v8h*)&sIn[lane * 16 + 8] = *(const v8h*)(latent_h + (size_t)g * LAT);
    __syncthreads();

    const int m = lane & 15;
    const int n = lane & 15;

    // A fragments: 16-wide input, K padded to 32 with zeros (matches D1f pad)
    v16h a[2];
    const int b0 = (lane >> 4) << 3;
    #pragma unroll
    for (int mt = 0; mt < 2; ++mt) {
        v8h lo = *(const v8h*)&sIn[(mt * 16 + m) * 16 + b0];
        v8h z  = {};
        a[mt] = __builtin_shufflevector(lo, z, 0,1,2,3,4,5,6,7,8,9,10,11,12,13,14,15);
    }

    // GEMM1: (32x16pad32)@(16x64) + bias, ReLU -> sH
    #pragma unroll
    for (int nt = 0; nt < 4; ++nt) {
        v16h b = load_frag(D1f, nt, lane);
        float bias = db1[nt * 16 + n];
        #pragma unroll
        for (int mt = 0; mt < 2; ++mt) {
            v8f c = {};
            c = __builtin_amdgcn_wmma_f32_16x16x32_f16(false, a[mt], false, b,
                                                       (short)0, c, false, false);
            #pragma unroll
            for (int r = 0; r < 8; ++r) {
                float v = c[r] + bias;
                sH[(mt * 16 + cd_m(lane, r)) * HID + nt * 16 + n] =
                    (_Float16)(v > 0.0f ? v : 0.0f);
            }
        }
    }
    __syncthreads();

    v16h a2[2][2];
    #pragma unroll
    for (int mt = 0; mt < 2; ++mt)
        #pragma unroll
        for (int ch = 0; ch < 2; ++ch)
            a2[mt][ch] = load_a32(&sH[(mt * 16 + m) * HID + ch * 32], lane);

    // GEMM2: (32x64)@(64x32) + bias -> out
    #pragma unroll
    for (int nt = 0; nt < 2; ++nt) {
        v8f c[2] = {{}, {}};
        #pragma unroll
        for (int ch = 0; ch < 2; ++ch) {
            v16h b = load_frag(D2f, nt * 2 + ch, lane);
            #pragma unroll
            for (int mt = 0; mt < 2; ++mt)
                c[mt] = __builtin_amdgcn_wmma_f32_16x16x32_f16(false, a2[mt][ch], false, b,
                                                               (short)0, c[mt], false, false);
        }
        float bias = db2[nt * 16 + n];
        #pragma unroll
        for (int mt = 0; mt < 2; ++mt)
            #pragma unroll
            for (int r = 0; r < 8; ++r)
                out[(size_t)(e0 + mt * 16 + cd_m(lane, r)) * EDGE_DIM + nt * 16 + n] =
                    c[mt][r] + bias;
    }
}

// --------------------------------------------------------------------------
extern "C" void kernel_launch(void* const* d_in, const int* in_sizes, int n_in,
                              void* d_out, int out_size, void* d_ws, size_t ws_size,
                              hipStream_t stream) {
    const int*   edge_index = (const int*)  d_in[0];
    const float* edge_attr  = (const float*)d_in[1];
    const float* node_emb   = (const float*)d_in[2];
    const float* ew_W1      = (const float*)d_in[3];
    const float* ew_b1      = (const float*)d_in[4];
    const float* ew_W2      = (const float*)d_in[5];
    const float* ew_b2      = (const float*)d_in[6];
    const float* root_W     = (const float*)d_in[7];
    const float* conv_b     = (const float*)d_in[8];
    const float* dec_W1     = (const float*)d_in[9];
    const float* dec_b1     = (const float*)d_in[10];
    const float* dec_W2     = (const float*)d_in[11];
    const float* dec_b2     = (const float*)d_in[12];
    float* out = (float*)d_out;

    const int E = in_sizes[1] / EDGE_DIM;      // 1,600,000
    const int N = in_sizes[2] / EMB;           // 100,000

    // workspace layout (fragment and agg offsets 1KB-aligned)
    char* ws = (char*)d_ws;
    _Float16* W1f = (_Float16*)ws;  ws += (size_t) 4 * 512 * 2;   // 4 KB
    _Float16* W2f = (_Float16*)ws;  ws += (size_t)16 * 512 * 2;   // 16 KB
    _Float16* D1f = (_Float16*)ws;  ws += (size_t) 4 * 512 * 2;   // 4 KB
    _Float16* D2f = (_Float16*)ws;  ws += (size_t) 4 * 512 * 2;   // 4 KB
    float*    agg = (float*)ws;     ws += (size_t)N * LAT * 4;    // 3.2 MB
    float*    cnt = (float*)ws;     ws += (size_t)N * 4;          // 0.4 MB
    _Float16* lat = (_Float16*)ws;                                // 1.6 MB

    const int nAgg = N * LAT;

    zero_kernel<<<(nAgg + 255) / 256, 256, 0, stream>>>(agg, cnt, nAgg, N);
    build_frags<<<(28 * 512 + 255) / 256, 256, 0, stream>>>(
        ew_W1, ew_W2, dec_W1, dec_W2, W1f, W2f, D1f, D2f);

    const int nTiles = E / EPT;   // E divisible by 32
    edge_msg_kernel<<<nTiles, 32, 0, stream>>>(
        edge_index, edge_attr, node_emb, W1f, ew_b1, W2f, ew_b2, agg, cnt, E);

    latent_kernel<<<(N + 255) / 256, 256, 0, stream>>>(
        node_emb, root_W, conv_b, agg, cnt, lat, N);

    decode_kernel<<<nTiles, 32, 0, stream>>>(
        edge_index, lat, D1f, dec_b1, D2f, dec_b2, out, E);
}
